// Light12_17197049053621
// MI455X (gfx1250) — matrix operand order repaired and checked by reading, not compiled, not run on hardware
//
#include <hip/hip_runtime.h>
#include <hip/hip_bf16.h>

// ======================================================================
// Light12 (GhostNet-ish SR net) for MI455X / gfx1250, wave32, WMMA.
// All 1x1 convs -> v_wmma_f32_16x16x32_f16 (f16 in, f32 accumulate).
// X tiles are DMA'd global->LDS with GLOBAL_LOAD_ASYNC_TO_LDS_B128
// (ASYNCcnt), then transpose-converted f32->f16 inside LDS.
// ======================================================================

typedef __attribute__((ext_vector_type(16))) _Float16 hvec16;
typedef __attribute__((ext_vector_type(8)))  float    fvec8;

#define ACT_NONE  0
#define ACT_RELU  1
#define ACT_PRELU 2

static __device__ __forceinline__ void wait_async0()
{
#if __has_builtin(__builtin_amdgcn_s_wait_asynccnt)
    __builtin_amdgcn_s_wait_asynccnt(0);
#else
    asm volatile("s_wait_asynccnt 0x0" ::: "memory");
#endif
}

// Issue one 16-byte async global->LDS copy from this lane.
// VDST operand = VGPR holding the 32-bit LDS byte address (low 32 bits of the
// generic pointer ARE the LDS address per ISA 10.2 aperture mapping).
static __device__ __forceinline__ void async_copy_b128(void* lds_dst, const void* gsrc)
{
    unsigned int       ldsa = (unsigned int)(__SIZE_TYPE__)lds_dst;
    unsigned long long ga   = (unsigned long long)(__SIZE_TYPE__)gsrc;
    asm volatile("global_load_async_to_lds_b128 %0, %1, off"
                 :: "v"(ldsa), "v"(ga) : "memory");
}

// ----------------------------------------------------------------------
// WMMA GEMM kernel for 1x1 conv over NCHW:
//   y[n, co_off+m, hw] = act( sum_k W[m,k] * x[n, cin_off+k, hw] + b[m] )
// Block = 128 threads (4 waves), each block handles 64 consecutive pixels
// (p = n*HW + hw; HW % 64 == 0 at every scale so n is tile-uniform).
// Each wave owns 16 pixels (B columns) and loops over all 16-row M tiles.
// A/B fragment gathers use contiguous k order (k0 + 16*hi + s), the same for both.
// ----------------------------------------------------------------------
__global__ __launch_bounds__(128)
void k_conv1x1_wmma(const float* __restrict__ x, int Cx, int cin_off,
                    float* __restrict__ y, int Cy, int co_off,
                    const float* __restrict__ w, const float* __restrict__ bias,
                    const float* __restrict__ slope,
                    int HW, int Cin, int Cout, int Kp, int Mp, int act)
{
    __shared__ float    Xstage[64][64];  // [k][pixel] raw f32 tile (async DMA target)
    __shared__ _Float16 Xs[64][64];      // [pixel][k] f16 (row stride 64 halves = 128B)
    __shared__ _Float16 Ws[128][64];     // [m][k]     f16
    __shared__ float    Bs[128];
    __shared__ float    Ss[128];

    const int tid = threadIdx.x;
    const int p0  = blockIdx.x * 64;
    const int n   = p0 / HW;            // uniform for whole tile
    const int hw0 = p0 - n * HW;

    __builtin_prefetch(w, 0, 3);        // gfx1250 global_prefetch_b8

    // ---- async DMA: Cin rows x 64 f32 pixels -> LDS staging [k][pixel] ----
    // Each segment = 16B (4 floats); row = 16 segments; base 16B-aligned
    // because hw0 % 64 == 0 and all tensors are 16B-aligned in the arena.
    {
        const int nseg = Cin * 16;
        for (int s = tid; s < nseg; s += 128) {
            int k = s >> 4, part = (s & 15) * 4;
            const float* gp = x + ((size_t)(n * Cx + cin_off + k)) * HW + hw0 + part;
            async_copy_b128(&Xstage[k][part], gp);
        }
    }

    // ---- stage weights meanwhile: zero-pad [Cout,Cin] -> [Mp,Kp], f32->f16 ----
    for (int i = tid; i < Mp * Kp; i += 128) {
        int m = i / Kp, k = i - m * Kp;
        float v = 0.0f;
        if (m < Cout && k < Cin) v = w[m * Cin + k];
        Ws[m][k] = (_Float16)v;
    }
    if (tid < Mp) {
        Bs[tid] = (tid < Cout) ? bias[tid] : 0.0f;
        Ss[tid] = (act == ACT_PRELU && tid < Cout) ? slope[tid] : 0.0f;
    }

    // ---- drain DMA, then LDS->LDS transpose + f32->f16 convert ----
    wait_async0();
    __syncthreads();
    for (int i = tid; i < 64 * Kp; i += 128) {
        int pp = i / Kp, k = i - pp * Kp;
        Xs[pp][k] = (_Float16)((k < Cin) ? Xstage[k][pp] : 0.0f);
    }
    __syncthreads();

    const int wv   = tid >> 5;
    const int lane = tid & 31;
    const int col  = lane & 15;        // N index / A-row index
    const int hi   = lane >> 4;        // half-wave selector
    const int px   = wv * 16 + col;    // pixel owned by this lane's B column

    for (int m0 = 0; m0 < Mp; m0 += 16) {
        fvec8 acc = {};
        const int mrow = m0 + col;     // A matrix: lanes 0-15 rows M=0..15, 16-31 same rows
        for (int k0 = 0; k0 < Kp; k0 += 32) {
            // A frag: lane holds K = k0+hi*16+{0..15}
            hvec16 a = *(const hvec16*)&Ws[mrow][k0 + hi * 16];
            // B frag: lanes 0-15 hold K=k0..k0+15, lanes 16-31 hold K=k0+16..k0+31
            hvec16 b = *(const hvec16*)&Xs[px][k0 + hi * 16];
            acc = __builtin_amdgcn_wmma_f32_16x16x32_f16(
                      false, a, false, b, (short)0, acc, false, false);
        }
        // D layout: VGPR r -> M = m0 + hi*8 + r, N = col
        const int mbase = m0 + hi * 8;
        #pragma unroll
        for (int r = 0; r < 8; ++r) {
            int mm = mbase + r;
            if (mm < Cout) {
                float v = acc[r] + Bs[mm];
                if (act == ACT_RELU)       v = fmaxf(v, 0.0f);
                else if (act == ACT_PRELU) { if (v < 0.0f) v *= Ss[mm]; }
                y[((size_t)(n * Cy + co_off + mm)) * HW + hw0 + px] = v;
            }
        }
    }
}

// ---------------- depthwise conv (k = 1 or 3, pad = k/2) ----------------
__global__ void k_dwconv(const float* __restrict__ x, int Cx, int cin_off,
                         float* __restrict__ y, int Cy, int co_off,
                         const float* __restrict__ w, const float* __restrict__ b,
                         int C, int H, int W, int ksz, int relu, int total)
{
    int idx = blockIdx.x * blockDim.x + threadIdx.x;
    if (idx >= total) return;
    int wp = idx % W; int t = idx / W;
    int h  = t % H;   t /= H;
    int c  = t % C;   int n = t / C;
    int pad = ksz >> 1;
    const float* xp = x + ((size_t)(n * Cx + cin_off + c)) * H * W;
    const float* wq = w + c * ksz * ksz;
    float acc = b[c];
    for (int i = 0; i < ksz; ++i) {
        int hh = h + i - pad; if ((unsigned)hh >= (unsigned)H) continue;
        for (int j = 0; j < ksz; ++j) {
            int ww = wp + j - pad; if ((unsigned)ww >= (unsigned)W) continue;
            acc += xp[hh * W + ww] * wq[i * ksz + j];
        }
    }
    if (relu) acc = fmaxf(acc, 0.0f);
    y[((size_t)(n * Cy + co_off + c)) * H * W + h * W + wp] = acc;
}

// ---------------- dense 3x3 conv (small Cin: head/dec/fire-e3) ----------------
__global__ void k_conv3x3(const float* __restrict__ x, int Cx, int cin_off,
                          float* __restrict__ y, int Cy, int co_off,
                          const float* __restrict__ w, const float* __restrict__ b,
                          const float* __restrict__ slope,
                          int Cin, int Cout, int H, int W, int act, int total)
{
    int idx = blockIdx.x * blockDim.x + threadIdx.x;
    if (idx >= total) return;
    int wp = idx % W; int t = idx / W;
    int h  = t % H;   t /= H;
    int co = t % Cout; int n = t / Cout;
    float acc = b[co];
    for (int ci = 0; ci < Cin; ++ci) {
        const float* xp = x + ((size_t)(n * Cx + cin_off + ci)) * H * W;
        const float* wq = w + ((size_t)(co * Cin + ci)) * 9;
        for (int i = 0; i < 3; ++i) {
            int hh = h + i - 1; if ((unsigned)hh >= (unsigned)H) continue;
            for (int j = 0; j < 3; ++j) {
                int ww = wp + j - 1; if ((unsigned)ww >= (unsigned)W) continue;
                acc += xp[hh * W + ww] * wq[i * 3 + j];
            }
        }
    }
    if (act == ACT_RELU)       acc = fmaxf(acc, 0.0f);
    else if (act == ACT_PRELU) { if (acc < 0.0f) acc *= slope[co]; }
    y[((size_t)(n * Cy + co_off + co)) * H * W + h * W + wp] = acc;
}

// -------- pixel_shuffle(r=2) with optional pre-PReLU (a1) / post-PReLU (a2) ----
__global__ void k_pshuffle(const float* __restrict__ x, int Cx, int cin_off,
                           const float* __restrict__ a1,
                           float* __restrict__ y, int Cy, int co_off,
                           const float* __restrict__ a2,
                           int Cout, int Ho, int Wo, int total)
{
    int idx = blockIdx.x * blockDim.x + threadIdx.x;
    if (idx >= total) return;
    int wo = idx % Wo; int t = idx / Wo;
    int ho = t % Ho;   t /= Ho;
    int c  = t % Cout; int n = t / Cout;
    int Hi = Ho >> 1, Wi = Wo >> 1;
    int cin = c * 4 + (ho & 1) * 2 + (wo & 1);
    float v = x[((size_t)(n * Cx + cin_off + cin)) * Hi * Wi + (ho >> 1) * Wi + (wo >> 1)];
    if (a1 && v < 0.0f) v *= a1[cin];
    if (a2 && v < 0.0f) v *= a2[c];
    y[((size_t)(n * Cy + co_off + c)) * Ho * Wo + ho * Wo + wo] = v;
}

// ---------------- elementwise residual add on channel views ----------------
__global__ void k_add(float* __restrict__ y, int Cy, int co_off,
                      const float* __restrict__ x, int Cx, int cin_off,
                      int C, int HW, int total)
{
    int idx = blockIdx.x * blockDim.x + threadIdx.x;
    if (idx >= total) return;
    int hw = idx % HW; int t = idx / HW;
    int c  = t % C;    int n = t / C;
    y[((size_t)(n * Cy + co_off + c)) * HW + hw] +=
        x[((size_t)(n * Cx + cin_off + c)) * HW + hw];
}

// ---------------- SE: global average pool (one block per (n,c)) ----------------
__global__ void k_avgpool(const float* __restrict__ x, int Cx, int cin_off,
                          float* __restrict__ mean, int C, int HW)
{
    int bidx = blockIdx.x;                 // = n*C + c
    int n = bidx / C, c = bidx % C;
    const float* xp = x + ((size_t)(n * Cx + cin_off + c)) * HW;
    __shared__ float red[256];
    float s = 0.0f;
    for (int i = threadIdx.x; i < HW; i += 256) s += xp[i];
    red[threadIdx.x] = s;
    __syncthreads();
    for (int st = 128; st > 0; st >>= 1) {
        if (threadIdx.x < st) red[threadIdx.x] += red[threadIdx.x + st];
        __syncthreads();
    }
    if (threadIdx.x == 0) mean[bidx] = red[0] / (float)HW;
}

// ---------------- SE: 2-layer MLP + hard-sigmoid gate (single block) ----------------
__global__ void k_segate(const float* __restrict__ mean,
                         const float* __restrict__ w1, const float* __restrict__ b1,
                         const float* __restrict__ w2, const float* __restrict__ b2,
                         float* __restrict__ gate, int N, int C, int R)
{
    __shared__ float s1[128];
    int tid = threadIdx.x;
    for (int i = tid; i < N * R; i += blockDim.x) {
        int n = i / R, j = i % R;
        float s = b1[j];
        for (int c = 0; c < C; ++c) s += w1[j * C + c] * mean[n * C + c];
        s1[i] = fmaxf(s, 0.0f);
    }
    __syncthreads();
    for (int i = tid; i < N * C; i += blockDim.x) {
        int n = i / C, c = i % C;
        float g = b2[c];
        for (int j = 0; j < R; ++j) g += w2[c * R + j] * s1[n * R + j];
        g = (g + 3.0f) * (1.0f / 6.0f);
        gate[i] = fminf(fmaxf(g, 0.0f), 1.0f);
    }
}

__global__ void k_sescale(float* __restrict__ y, int Cy, int co_off,
                          const float* __restrict__ gate, int C, int HW, int total)
{
    int idx = blockIdx.x * blockDim.x + threadIdx.x;
    if (idx >= total) return;
    int hw = idx % HW; int t = idx / HW;
    int c  = t % C;    int n = t / C;
    y[((size_t)(n * Cy + co_off + c)) * HW + hw] *= gate[n * C + c];
}

// ------- final: out = feat + pixel_conv(feat, z) with 5x5 adaptive kernel -------
// kernel channel j = kw*5 + kh (reference permutes (kw,kh)); feat zero-padded by 2.
__global__ void k_pixconv(const float* __restrict__ feat, const float* __restrict__ z,
                          float* __restrict__ out, int H, int W, int total)
{
    int idx = blockIdx.x * blockDim.x + threadIdx.x;
    if (idx >= total) return;
    int wp = idx % W; int t = idx / W;
    int h  = t % H;   int n = t / H;
    size_t HW = (size_t)H * W;
    const float* fp = feat + (size_t)n * HW;
    float acc = fp[h * W + wp];
    for (int kh = 0; kh < 5; ++kh) {
        int hh = h + kh - 2; if ((unsigned)hh >= (unsigned)H) continue;
        for (int kw = 0; kw < 5; ++kw) {
            int ww = wp + kw - 2; if ((unsigned)ww >= (unsigned)W) continue;
            float kv = z[((size_t)(n * 25) + kw * 5 + kh) * HW + h * W + wp];
            acc += fp[hh * W + ww] * kv;
        }
    }
    out[(size_t)n * HW + h * W + wp] = acc;
}

// ======================================================================
// Host-side orchestration
// ======================================================================
static inline int cdiv(int a, int b) { return (a + b - 1) / b; }

static void conv1x1(hipStream_t st, const float* x, int Cx, int cin_off,
                    float* y, int Cy, int co_off,
                    const float* w, const float* b, const float* slope, int act,
                    int N, int HW, int Cin, int Cout)
{
    int Kp = (Cin  + 31) & ~31;   // 32 or 64
    int Mp = (Cout + 15) & ~15;   // <= 128
    int blocks = (N * HW) / 64;
    k_conv1x1_wmma<<<blocks, 128, 0, st>>>(x, Cx, cin_off, y, Cy, co_off,
                                           w, b, slope, HW, Cin, Cout, Kp, Mp, act);
}

static void dwconv(hipStream_t st, const float* x, int Cx, int cin_off,
                   float* y, int Cy, int co_off,
                   const float* w, const float* b,
                   int N, int C, int H, int W, int ksz, int relu)
{
    int total = N * C * H * W;
    k_dwconv<<<cdiv(total, 256), 256, 0, st>>>(x, Cx, cin_off, y, Cy, co_off,
                                               w, b, C, H, W, ksz, relu, total);
}

// ghost_bottleneck: out (dense oup channels) = g2(SE?(g1(x))) + shortcut(x)
static void run_gb(hipStream_t st, const float* x, int Cx, int cin_off,
                   float* out, int N, int H, int W,
                   int inp, int mid, int oup, int ksc,
                   const float* g1w1, const float* g1b1, const float* g1w2, const float* g1b2,
                   const float* g2w1, const float* g2b1, const float* g2w2, const float* g2b2,
                   const float* sew1, const float* seb1, const float* sew2, const float* seb2,
                   const float* scdw, const float* scdb, const float* scpw, const float* scpb,
                   float* T1, float* T3, float* T4, float* mean, float* gate)
{
    const int HW = H * W;
    const int ic1 = mid / 2, ic2 = oup / 2;
    // ghost module 1 (ReLU): y1 = conv1x1, y2 = dw3x3(y1), concat in T1[mid]
    conv1x1(st, x, Cx, cin_off, T1, mid, 0, g1w1, g1b1, nullptr, ACT_RELU, N, HW, inp, ic1);
    dwconv(st, T1, mid, 0, T1, mid, ic1, g1w2, g1b2, N, ic1, H, W, 3, 1);
    if (sew1) {   // squeeze-excite with hard sigmoid
        k_avgpool<<<N * mid, 256, 0, st>>>(T1, mid, 0, mean, mid, HW);
        k_segate<<<1, 256, 0, st>>>(mean, sew1, seb1, sew2, seb2, gate, N, mid, mid / 4);
        int tot = N * mid * HW;
        k_sescale<<<cdiv(tot, 256), 256, 0, st>>>(T1, mid, 0, gate, mid, HW, tot);
    }
    // ghost module 2 (linear) -> out[oup]
    conv1x1(st, T1, mid, 0, out, oup, 0, g2w1, g2b1, nullptr, ACT_NONE, N, HW, mid, ic2);
    dwconv(st, out, oup, 0, out, oup, ic2, g2w2, g2b2, N, ic2, H, W, 3, 0);
    // shortcut
    int tot = N * oup * HW;
    if (scdw) {
        dwconv(st, x, Cx, cin_off, T3, inp, 0, scdw, scdb, N, inp, H, W, ksc, 0);
        conv1x1(st, T3, inp, 0, T4, oup, 0, scpw, scpb, nullptr, ACT_NONE, N, HW, inp, oup);
        k_add<<<cdiv(tot, 256), 256, 0, st>>>(out, oup, 0, T4, oup, 0, oup, HW, tot);
    } else {     // identity (inp == oup)
        k_add<<<cdiv(tot, 256), 256, 0, st>>>(out, oup, 0, x, Cx, cin_off, oup, HW, tot);
    }
}

extern "C" void kernel_launch(void* const* d_in, const int* in_sizes, int n_in,
                              void* d_out, int out_size, void* d_ws, size_t ws_size,
                              hipStream_t stream)
{
    (void)in_sizes; (void)n_in; (void)out_size; (void)ws_size;
    // Input order: jax pytree leaves of {"params": {...}, "x": ...} with dict keys
    // sorted alphabetically at every level ("params" < "x"); gb dicts flatten as
    // g1(b1,b2,w1,w2), g2(...), sc_db, sc_dw, sc_pb, sc_pw, se_b1, se_b2, se_w1, se_w2.
    auto F = [&](int i) { return (const float*)d_in[i]; };
    // params (leaf indices)
    const float *dec_a = F(0), *dec_b = F(1), *dec_w = F(2);
    const float *eesp_a = F(3), *eesp_cb = F(4), *eesp_cw = F(5);
    // eesp_gb1 = gb(64,64,16, k=1, no SE, has SC): 6..17
    // eesp_gb2 = gb(16,16,16, no SE, no SC): 18..25
    // fire: 26 e1b, 27 e1w, 28 e3b, 29 e3w, 30 sb, 31 sw
    const float *f_e1b = F(26), *f_e1w = F(27), *f_e3b = F(28), *f_e3w = F(29),
                *f_sb = F(30), *f_sw = F(31);
    const float *head_a = F(32), *head_b = F(33), *head_w = F(34);
    const float *up11_a1 = F(35), *up11_a2 = F(36);     // up11_gb: 37..52
    const float *up12_a1 = F(53), *up12_a2 = F(54);     // up12_gb: 55..70
    const float *up21_a1 = F(71), *up21_a2 = F(72);     // up21_gb: 73..88
    const float *up22_a2 = F(89);                       // up22_gb: 90..101
    const float *x_in = F(102);                         // (4,1,192,192)

    const int N = 4, H0 = 192, W0 = 192;
    const int H1 = 384, W1 = 384, H2 = 768, W2 = 768;
    const int P0 = N * H0 * W0, P1 = N * H1 * W1, P2 = N * H2 * W2;

    // ---- workspace arena (floats) ----
    float* ws = (float*)d_ws;
    size_t cur = 0;
    auto A = [&](size_t nf) { float* p = ws + cur; cur += nf; return p; };
    float* head    = A((size_t)64 * P0);
    float* t16a    = A((size_t)16 * P0);
    float* t16b    = A((size_t)16 * P0);
    float* x1      = A((size_t)64 * P0);
    float* up11pre = A((size_t)128 * P0);
    float* up12pre = A((size_t)128 * P0);
    float* cat1    = A((size_t)64 * P1);   // [0,32)=up11ps, [32,64)=up12ps
    float* up21pre = A((size_t)64 * P1);
    float* up22pre = A((size_t)64 * P1);
    float* cat2    = A((size_t)32 * P2);   // [0,16)=up21ps, [16,32)=up22ps
    float* feat    = A((size_t)P2);
    float* fs      = A((size_t)4 * P2);
    float* fz      = A((size_t)25 * P2);
    float* T1      = A((size_t)64 * P1);   // gb scratch: g1 output (mid ch)
    float* T3      = A((size_t)64 * P1);   // gb scratch: sc_dw output (inp ch)
    float* T4      = A((size_t)64 * P1);   // gb scratch: sc_pw output (oup ch)
    float* mean    = A(1024);
    float* gate    = A(1024);

    // ---- head: prelu(conv3x3(x, 1->64)) ----
    {
        int tot = N * 64 * H0 * W0;
        k_conv3x3<<<cdiv(tot, 256), 256, 0, stream>>>(x_in, 1, 0, head, 64, 0,
            head_w, head_b, head_a, 1, 64, H0, W0, ACT_PRELU, tot);
    }

    // ---- up11 = GB(head; 64,64,128, SE, SC k=3) then PS2 with a1/a2 -> cat1[0:32) ----
    run_gb(stream, head, 64, 0, up11pre, N, H0, W0, 64, 64, 128, 3,
           F(39), F(37), F(40), F(38),   // g1 w1,b1,w2,b2
           F(43), F(41), F(44), F(42),   // g2
           F(51), F(49), F(52), F(50),   // se w1,b1,w2,b2
           F(46), F(45), F(48), F(47),   // sc dw,db,pw,pb
           T1, T3, T4, mean, gate);
    {
        int tot = N * 32 * H1 * W1;
        k_pshuffle<<<cdiv(tot, 256), 256, 0, stream>>>(up11pre, 128, 0, up11_a1,
            cat1, 64, 0, up11_a2, 32, H1, W1, tot);
    }

    // ---- eesp branch ----
    run_gb(stream, head, 64, 0, t16a, N, H0, W0, 64, 64, 16, 1,
           F(8), F(6), F(9), F(7),       // g1
           F(12), F(10), F(13), F(11),   // g2
           nullptr, nullptr, nullptr, nullptr,
           F(15), F(14), F(17), F(16),   // sc (k=1)
           T1, T3, T4, mean, gate);
    run_gb(stream, t16a, 16, 0, t16b, N, H0, W0, 16, 16, 16, 3,
           F(20), F(18), F(21), F(19),   // g1
           F(24), F(22), F(25), F(23),   // g2
           nullptr, nullptr, nullptr, nullptr,
           nullptr, nullptr, nullptr, nullptr,  // identity shortcut
           T1, T3, T4, mean, gate);
    conv1x1(stream, t16b, 16, 0, x1, 64, 0, eesp_cw, eesp_cb, eesp_a, ACT_PRELU,
            N, H0 * W0, 16, 64);

    // ---- up12 = GB(x1; 64,64,128, SE, SC) then PS2 -> cat1[32:64) ----
    run_gb(stream, x1, 64, 0, up12pre, N, H0, W0, 64, 64, 128, 3,
           F(57), F(55), F(58), F(56),
           F(61), F(59), F(62), F(60),
           F(69), F(67), F(70), F(68),
           F(64), F(63), F(66), F(65),
           T1, T3, T4, mean, gate);
    {
        int tot = N * 32 * H1 * W1;
        k_pshuffle<<<cdiv(tot, 256), 256, 0, stream>>>(up12pre, 128, 0, up12_a1,
            cat1, 64, 32, up12_a2, 32, H1, W1, tot);
    }

    // ---- up21 = GB(cat1[0:32); 32,32,64, SE, SC) then PS2 -> cat2[0:16) ----
    run_gb(stream, cat1, 64, 0, up21pre, N, H1, W1, 32, 32, 64, 3,
           F(75), F(73), F(76), F(74),
           F(79), F(77), F(80), F(78),
           F(87), F(85), F(88), F(86),
           F(82), F(81), F(84), F(83),
           T1, T3, T4, mean, gate);
    {
        int tot = N * 16 * H2 * W2;
        k_pshuffle<<<cdiv(tot, 256), 256, 0, stream>>>(up21pre, 64, 0, up21_a1,
            cat2, 32, 0, up21_a2, 16, H2, W2, tot);
    }

    // ---- up22 = GB(cat1; 64,64,64, SE, identity) then PS2 (a2 only) -> cat2[16:32) ----
    run_gb(stream, cat1, 64, 0, up22pre, N, H1, W1, 64, 64, 64, 3,
           F(92), F(90), F(93), F(91),
           F(96), F(94), F(97), F(95),
           F(100), F(98), F(101), F(99),
           nullptr, nullptr, nullptr, nullptr,
           T1, T3, T4, mean, gate);
    {
        int tot = N * 16 * H2 * W2;
        k_pshuffle<<<cdiv(tot, 256), 256, 0, stream>>>(up22pre, 64, 0, nullptr,
            cat2, 32, 16, up22_a2, 16, H2, W2, tot);
    }

    // ---- decoder: feat = prelu(conv3x3(cat2, 32->1)) ----
    {
        int tot = N * 1 * H2 * W2;
        k_conv3x3<<<cdiv(tot, 256), 256, 0, stream>>>(cat2, 32, 0, feat, 1, 0,
            dec_w, dec_b, dec_a, 32, 1, H2, W2, ACT_PRELU, tot);
    }

    // ---- fire: s = relu(1x1 32->4); z = relu([e1(1x1 4->4); e3(3x3 4->21)]) ----
    conv1x1(stream, cat2, 32, 0, fs, 4, 0, f_sw, f_sb, nullptr, ACT_RELU, N, H2 * W2, 32, 4);
    conv1x1(stream, fs, 4, 0, fz, 25, 0, f_e1w, f_e1b, nullptr, ACT_RELU, N, H2 * W2, 4, 4);
    {
        int tot = N * 21 * H2 * W2;
        k_conv3x3<<<cdiv(tot, 256), 256, 0, stream>>>(fs, 4, 0, fz, 25, 4,
            f_e3w, f_e3b, nullptr, 4, 21, H2, W2, ACT_RELU, tot);
    }

    // ---- final: out = feat + pixel_conv(feat, z) ----
    {
        int tot = N * H2 * W2;
        k_pixconv<<<cdiv(tot, 256), 256, 0, stream>>>(feat, fz, (float*)d_out, H2, W2, tot);
    }
}
